// GATModel_10617159155782
// MI455X (gfx1250) — matrix operand order repaired and checked by reading, not compiled, not run on hardware
//
#include <hip/hip_runtime.h>
#include <hip/hip_bf16.h>

// ---------------------------------------------------------------------------
// GAT (3 layers, H=8, C=64) for gfx1250.
// GEMMs use v_wmma_f32_16x16x32_f16 (wave32); each wave computes a 16x64
// strip of C (4 accumulators) so one A fragment feeds 4 WMMAs per K-step.
// ---------------------------------------------------------------------------

typedef __attribute__((ext_vector_type(16))) _Float16 v16h;
typedef __attribute__((ext_vector_type(8)))  float    v8f;

#define HEADS 8
#define HID   64
#define HC    512
#define FIN   18
#define FOUT  15
#define NEG_SLOPE 0.2f
#define EPS_F 1e-16f

// ---------------- helpers ----------------

__device__ __forceinline__ unsigned int encodeMaxKey(float f) {
  unsigned int u = __float_as_uint(f);
  return (u & 0x80000000u) ? ~u : (u | 0x80000000u);
}
__device__ __forceinline__ float decodeMaxKey(unsigned int k) {
  unsigned int u = (k & 0x80000000u) ? (k & 0x7FFFFFFFu) : ~k;
  return __uint_as_float(u);
}

// ---------------- init kernels ----------------

__global__ void k_init_f32(float* __restrict__ p, float v, int n) {
  int i = blockIdx.x * blockDim.x + threadIdx.x;
  if (i < n) p[i] = v;
}
__global__ void k_init_u32(unsigned int* __restrict__ p, unsigned int v, int n) {
  int i = blockIdx.x * blockDim.x + threadIdx.x;
  if (i < n) p[i] = v;
}

// ---------------- f32 -> f16 convert with K padding ----------------
__global__ void k_cvt_pad(const float* __restrict__ src, _Float16* __restrict__ dst,
                          int rows, int scols, int dcols) {
  int i = blockIdx.x * blockDim.x + threadIdx.x;
  int total = rows * dcols;
  if (i >= total) return;
  int r = i / dcols, c = i - r * dcols;
  float v = (c < scols) ? src[(size_t)r * scols + c] : 0.0f;
  dst[i] = (_Float16)v;
}

// ---------------- W [K, Nc] f32 -> Wt [Nc, Kpad] f16 ----------------
__global__ void k_transpose_w(const float* __restrict__ W, _Float16* __restrict__ Wt,
                              int K, int Nc, int Kpad) {
  int i = blockIdx.x * blockDim.x + threadIdx.x;
  int total = Nc * Kpad;
  if (i >= total) return;
  int n = i / Kpad, k = i - n * Kpad;
  float v = (k < K) ? W[(size_t)k * Nc + n] : 0.0f;
  Wt[i] = (_Float16)v;
}

// ---------------- WMMA GEMM: C[M,Nc] = A[M,Kp] * Bt[Nc,Kp]^T ----------------
// One wave computes a 16x64 strip: 1 A fragment -> 4 independent WMMAs/K-step.
union FragA { v16h v; _Float16 e[16]; };

__global__ void __launch_bounds__(256)
k_wmma_gemm(const _Float16* __restrict__ A,   // [M, Kp] row-major
            const _Float16* __restrict__ Bt,  // [Nc, Kp] row-major (W transposed)
            float* __restrict__ C,            // [M, Nc]
            int M, int Kp, int Nc) {
  const int lane = threadIdx.x & 31;
  const int wave = threadIdx.x >> 5;
  const int stripsN = Nc >> 6;                       // 64-wide strips
  const int tile = blockIdx.x * (blockDim.x >> 5) + wave;
  const int totalTiles = ((M + 15) >> 4) * stripsN;
  if (tile >= totalTiles) return;
  const int tm = tile / stripsN;
  const int tn = (tile - tm * stripsN) << 2;         // first 16-col tile of strip

  const int r  = lane & 15;
  const int hi = lane >> 4;

  int arowIdx = tm * 16 + r;
  if (arowIdx >= M) arowIdx = M - 1;                 // clamp (masked at store)
  const _Float16* arow = A + (size_t)arowIdx * Kp;
  const _Float16* brow[4];
#pragma unroll
  for (int s = 0; s < 4; ++s)
    brow[s] = Bt + (size_t)((tn + s) * 16 + r) * Kp;

  v8f acc0 = {0.f,0.f,0.f,0.f,0.f,0.f,0.f,0.f};
  v8f acc1 = acc0, acc2 = acc0, acc3 = acc0;

  for (int k0 = 0; k0 < Kp; k0 += 32) {
    FragA a, b0, b1, b2, b3;
    // A 16x32 f16 layout: VGPR j holds K = (j<4?0:16) + hi*8 + (j&3)*2 (+1)
#pragma unroll
    for (int j = 0; j < 8; ++j) {
      int ka = k0 + ((j < 4) ? 0 : 16) + hi * 8 + (j & 3) * 2;
      a.e[2 * j]     = arow[ka];
      a.e[2 * j + 1] = arow[ka + 1];
      // B 32x16 f16 layout: N = lane&15, VGPR j holds K = (j<4?0:16) + hi*8 + (j&3)*2 (+1)
      int kb = k0 + ((j < 4) ? 0 : 16) + hi * 8 + (j & 3) * 2;
      b0.e[2 * j] = brow[0][kb]; b0.e[2 * j + 1] = brow[0][kb + 1];
      b1.e[2 * j] = brow[1][kb]; b1.e[2 * j + 1] = brow[1][kb + 1];
      b2.e[2 * j] = brow[2][kb]; b2.e[2 * j + 1] = brow[2][kb + 1];
      b3.e[2 * j] = brow[3][kb]; b3.e[2 * j + 1] = brow[3][kb + 1];
    }
    acc0 = __builtin_amdgcn_wmma_f32_16x16x32_f16(false, a.v, false, b0.v, (short)0, acc0, false, false);
    acc1 = __builtin_amdgcn_wmma_f32_16x16x32_f16(false, a.v, false, b1.v, (short)0, acc1, false, false);
    acc2 = __builtin_amdgcn_wmma_f32_16x16x32_f16(false, a.v, false, b2.v, (short)0, acc2, false, false);
    acc3 = __builtin_amdgcn_wmma_f32_16x16x32_f16(false, a.v, false, b3.v, (short)0, acc3, false, false);
  }

  // C/D layout: VGPR q -> M = q + hi*8, N = lane&15
  const int cmBase = tm * 16 + hi * 8;
  const int cnBase = (lane & 15);
  v8f accs[4] = {acc0, acc1, acc2, acc3};
#pragma unroll
  for (int s = 0; s < 4; ++s) {
    const int cn = (tn + s) * 16 + cnBase;
#pragma unroll
    for (int q = 0; q < 8; ++q) {
      int cm = cmBase + q;
      if (cm < M) C[(size_t)cm * Nc + cn] = accs[s][q];
    }
  }
}

// ---------------- per-(node, head) attention logits ----------------
__global__ void k_alphas(const float* __restrict__ h,
                         const float* __restrict__ a_src, const float* __restrict__ a_dst,
                         float* __restrict__ as_, float* __restrict__ ad_, int N) {
  int i = blockIdx.x * blockDim.x + threadIdx.x;
  if (i >= N * HEADS) return;
  int n = i >> 3, hd = i & 7;
  const float* hp = h + (size_t)n * HC + hd * HID;
  const float* asrc = a_src + hd * HID;
  const float* adst = a_dst + hd * HID;
  float ss = 0.f, sd = 0.f;
#pragma unroll 8
  for (int c = 0; c < HID; ++c) {
    float v = hp[c];
    ss += v * asrc[c];
    sd += v * adst[c];
  }
  as_[i] = ss;
  ad_[i] = sd;
}

// ---------------- edge pass 1: segment max (monotone-uint atomicMax) --------
__global__ void k_edge_max(const int* __restrict__ eidx, int E, int N,
                           const float* __restrict__ as_, const float* __restrict__ ad_,
                           unsigned int* __restrict__ mmax) {
  int i = blockIdx.x * blockDim.x + threadIdx.x;
  int Etot = E + N;
  if (i >= Etot * HEADS) return;
  int e = i >> 3, hd = i & 7;
  int s, d;
  if (e < E) { s = eidx[e]; d = eidx[E + e]; } else { s = d = e - E; }
  float v = as_[s * HEADS + hd] + ad_[d * HEADS + hd];
  v = (v > 0.f) ? v : NEG_SLOPE * v;
  atomicMax(&mmax[d * HEADS + hd], encodeMaxKey(v));
}

// ---------------- edge pass 2: segment sum of exp(e - m) ----------------
__global__ void k_edge_sum(const int* __restrict__ eidx, int E, int N,
                           const float* __restrict__ as_, const float* __restrict__ ad_,
                           const unsigned int* __restrict__ mmax,
                           float* __restrict__ denom) {
  int i = blockIdx.x * blockDim.x + threadIdx.x;
  int Etot = E + N;
  if (i >= Etot * HEADS) return;
  int e = i >> 3, hd = i & 7;
  int s, d;
  if (e < E) { s = eidx[e]; d = eidx[E + e]; } else { s = d = e - E; }
  float v = as_[s * HEADS + hd] + ad_[d * HEADS + hd];
  v = (v > 0.f) ? v : NEG_SLOPE * v;
  float m = decodeMaxKey(mmax[d * HEADS + hd]);
  atomicAdd(&denom[d * HEADS + hd], __expf(v - m));
}

// ---------------- edge pass 3: weighted message scatter-add ----------------
// 128 threads per edge; each thread handles 4 channels (one head per thread).
__global__ void k_edge_msg(const int* __restrict__ eidx, int E, int N,
                           const float* __restrict__ h,
                           const float* __restrict__ as_, const float* __restrict__ ad_,
                           const unsigned int* __restrict__ mmax,
                           const float* __restrict__ denom,
                           float* __restrict__ out) {
  long long gid = (long long)blockIdx.x * blockDim.x + threadIdx.x;
  long long Etot = (long long)E + N;
  if (gid >= Etot * 128) return;
  int t = (int)(gid & 127);
  int e = (int)(gid >> 7);
  int s, d;
  if (e < E) { s = eidx[e]; d = eidx[E + e]; } else { s = d = e - E; }
  int hd = t >> 4;                 // 16 threads per head (16*4 = 64 channels)
  int c0 = t * 4;                  // global channel [0, 512)
  float v = as_[s * HEADS + hd] + ad_[d * HEADS + hd];
  v = (v > 0.f) ? v : NEG_SLOPE * v;
  float m = decodeMaxKey(mmax[d * HEADS + hd]);
  float w = __expf(v - m) / (denom[d * HEADS + hd] + EPS_F);
  const float4 hv = *reinterpret_cast<const float4*>(h + (size_t)s * HC + c0);
  float* op = out + (size_t)d * HC + c0;
  atomicAdd(op + 0, hv.x * w);
  atomicAdd(op + 1, hv.y * w);
  atomicAdd(op + 2, hv.z * w);
  atomicAdd(op + 3, hv.w * w);
}

// ---------------- bias + optional ELU (in place on acc) ----------------
__global__ void k_bias_act(float* __restrict__ acc, const float* __restrict__ b,
                           int N, int doElu) {
  int i = blockIdx.x * blockDim.x + threadIdx.x;
  if (i >= N * HC) return;
  int c = i & (HC - 1);
  float v = acc[i] + b[c];
  if (doElu) v = (v > 0.f) ? v : (__expf(v) - 1.0f);
  acc[i] = v;
}

// ---------------- head mean + bias: [N,8,64] -> [N,64] ----------------
__global__ void k_mean_heads(const float* __restrict__ acc, const float* __restrict__ b3,
                             float* __restrict__ x3, int N) {
  int i = blockIdx.x * blockDim.x + threadIdx.x;
  if (i >= N * HID) return;
  int n = i >> 6, c = i & 63;
  const float* p = acc + (size_t)n * HC + c;
  float s = 0.f;
#pragma unroll
  for (int hd = 0; hd < HEADS; ++hd) s += p[hd * HID];
  x3[i] = s * (1.0f / HEADS) + b3[c];
}

// ---------------- final tiny GEMM: [N,64] @ [64,15] + b ----------------
__global__ void k_final(const float* __restrict__ x3, const float* __restrict__ Wout,
                        const float* __restrict__ bout, float* __restrict__ out, int N) {
  int i = blockIdx.x * blockDim.x + threadIdx.x;
  if (i >= N * FOUT) return;
  int n = i / FOUT, j = i - n * FOUT;
  const float* xp = x3 + (size_t)n * HID;
  float s = bout[j];
#pragma unroll 8
  for (int c = 0; c < HID; ++c) s += xp[c] * Wout[c * FOUT + j];
  out[i] = s;
}

// ---------------------------------------------------------------------------

static inline int ceil_div(long long a, long long b) { return (int)((a + b - 1) / b); }

extern "C" void kernel_launch(void* const* d_in, const int* in_sizes, int n_in,
                              void* d_out, int out_size, void* d_ws, size_t ws_size,
                              hipStream_t stream) {
  const float* x      = (const float*)d_in[0];
  const int*   eidx   = (const int*)d_in[1];
  const float* W1     = (const float*)d_in[2];
  const float* a1s    = (const float*)d_in[3];
  const float* a1d    = (const float*)d_in[4];
  const float* b1     = (const float*)d_in[5];
  const float* W2     = (const float*)d_in[6];
  const float* a2s    = (const float*)d_in[7];
  const float* a2d    = (const float*)d_in[8];
  const float* b2     = (const float*)d_in[9];
  const float* W3     = (const float*)d_in[10];
  const float* a3s    = (const float*)d_in[11];
  const float* a3d    = (const float*)d_in[12];
  const float* b3     = (const float*)d_in[13];
  const float* Wout   = (const float*)d_in[14];
  const float* bout   = (const float*)d_in[15];
  float* out = (float*)d_out;

  const int N = in_sizes[0] / FIN;
  const int E = in_sizes[1] / 2;
  const long long Etot = (long long)E + N;

  // ---- workspace carve ----
  char* base = (char*)d_ws;
  size_t off = 0;
  auto carve = [&](size_t bytes) -> char* {
    char* p = base + off;
    off = (off + bytes + 255) & ~(size_t)255;
    return p;
  };
  float*        hf32  = (float*)        carve((size_t)N * HC * sizeof(float));
  float*        acc   = (float*)        carve((size_t)N * HC * sizeof(float));
  _Float16*     af16  = (_Float16*)     carve((size_t)N * HC * sizeof(_Float16));
  _Float16*     wt16  = (_Float16*)     carve((size_t)HC * HC * sizeof(_Float16));
  float*        as_   = (float*)        carve((size_t)N * HEADS * sizeof(float));
  float*        ad_   = (float*)        carve((size_t)N * HEADS * sizeof(float));
  unsigned int* mmax  = (unsigned int*) carve((size_t)N * HEADS * sizeof(unsigned int));
  float*        denom = (float*)        carve((size_t)N * HEADS * sizeof(float));
  float*        x3    = (float*)        carve((size_t)N * HID * sizeof(float));
  (void)ws_size;

  const int TB = 256;
  const int nNH    = N * HEADS;
  const int nNHC   = N * HC;
  const int nEdgeH = (int)(Etot * HEADS);

  auto run_layer = [&](const float* aSrcVec, const float* aDstVec, const float* bias,
                       int doElu) {
    k_alphas<<<ceil_div(nNH, TB), TB, 0, stream>>>(hf32, aSrcVec, aDstVec, as_, ad_, N);
    k_init_u32<<<ceil_div(nNH, TB), TB, 0, stream>>>(mmax, 0u, nNH);
    k_init_f32<<<ceil_div(nNH, TB), TB, 0, stream>>>(denom, 0.f, nNH);
    k_init_f32<<<ceil_div(nNHC, TB), TB, 0, stream>>>(acc, 0.f, nNHC);
    k_edge_max<<<ceil_div(nEdgeH, TB), TB, 0, stream>>>(eidx, E, N, as_, ad_, mmax);
    k_edge_sum<<<ceil_div(nEdgeH, TB), TB, 0, stream>>>(eidx, E, N, as_, ad_, mmax, denom);
    k_edge_msg<<<ceil_div(Etot * 128, TB), TB, 0, stream>>>(eidx, E, N, hf32, as_, ad_,
                                                            mmax, denom, acc);
    if (doElu >= 0)
      k_bias_act<<<ceil_div(nNHC, TB), TB, 0, stream>>>(acc, bias, N, doElu);
  };

  // one wave per 16x64 strip
  const int gemmTiles  = ((N + 15) / 16) * (HC / 64);
  const int gemmBlocks = ceil_div(gemmTiles, TB / 32);

  // ---------------- Layer 1: K = 18 padded to 32 ----------------
  k_cvt_pad<<<ceil_div((long long)N * 32, TB), TB, 0, stream>>>(x, af16, N, FIN, 32);
  k_transpose_w<<<ceil_div(HC * 32, TB), TB, 0, stream>>>(W1, wt16, FIN, HC, 32);
  k_wmma_gemm<<<gemmBlocks, TB, 0, stream>>>(af16, wt16, hf32, N, 32, HC);
  run_layer(a1s, a1d, b1, /*elu=*/1);

  // ---------------- Layer 2: K = 512 ----------------
  k_cvt_pad<<<ceil_div((long long)nNHC, TB), TB, 0, stream>>>(acc, af16, N, HC, HC);
  k_transpose_w<<<ceil_div(HC * HC, TB), TB, 0, stream>>>(W2, wt16, HC, HC, HC);
  k_wmma_gemm<<<gemmBlocks, TB, 0, stream>>>(af16, wt16, hf32, N, HC, HC);
  run_layer(a2s, a2d, b2, /*elu=*/1);

  // ---------------- Layer 3: K = 512, mean over heads ----------------
  k_cvt_pad<<<ceil_div((long long)nNHC, TB), TB, 0, stream>>>(acc, af16, N, HC, HC);
  k_transpose_w<<<ceil_div(HC * HC, TB), TB, 0, stream>>>(W3, wt16, HC, HC, HC);
  k_wmma_gemm<<<gemmBlocks, TB, 0, stream>>>(af16, wt16, hf32, N, HC, HC);
  run_layer(a3s, a3d, nullptr, /*elu=*/-1);  // no bias/elu here
  k_mean_heads<<<ceil_div(N * HID, TB), TB, 0, stream>>>(acc, b3, x3, N);

  // ---------------- Output projection ----------------
  k_final<<<ceil_div(N * FOUT, TB), TB, 0, stream>>>(x3, Wout, bout, out, N);
}
